// HeteroGCN_21182778704707
// MI455X (gfx1250) — matrix operand order repaired and checked by reading, not compiled, not run on hardware
//
#include <hip/hip_runtime.h>
#include <hip/hip_bf16.h>

// ---------------------------------------------------------------------------
// HeteroGCN forward for MI455X (gfx1250, wave32, WMMA + TDM).
//   GEMMs   -> v_wmma_f32_16x16x32_bf16 (bf16 in, f32 accum); weights are
//              pre-swizzled into contiguous-k (k = 16*half + e) B-fragment lane order so each
//              fragment is one aligned 32B LDS read.
//   Staging -> Tensor Data Mover (tensor_load_to_lds + s_wait_tensorcnt) when
//              the builtin exists; cooperative 128b copy otherwise.
//   Scatter -> edge-parallel global_atomic_add_f32 (L2-resident: working set
//              ~170MB < 192MB L2).
//   BN      -> two-stage sum/sumsq reduction + fused normalize/LeakyReLU.
// ---------------------------------------------------------------------------

#define NB  131072
#define NG  32768
#define EBB 2097152
#define EGB 65536
#define EBG 65536
#define CIN 32
#define CC  64
#define COUT 32
#define NL  2
#define NT  5

#ifndef USE_TDM
#define USE_TDM 1
#endif

typedef __attribute__((ext_vector_type(16))) __bf16        v16bf;
typedef __attribute__((ext_vector_type(8)))  float         v8f;
typedef __attribute__((ext_vector_type(4)))  unsigned int  v4u;
typedef __attribute__((ext_vector_type(8)))  int           v8i;
typedef __attribute__((ext_vector_type(4)))  int           v4i;

// ---------------------------------------------------------------------------
// Pre-swizzle weights (f32 row-major [KD,NOUT]) into B-fragment order (bf16):
//   dst[((c*NTILE + t)*32 + lane)*16 + e] = W[(c*32 + (lane>>4)*16 + e)*NOUT
//                                             + t*16 + (lane&15)]
// so a lane's 16 bf16 values for (chunk c, tile t) are 32 contiguous bytes.
// ---------------------------------------------------------------------------
template <int KD, int NOUT>
__global__ __launch_bounds__(256)
void swizzle_w(const float* __restrict__ W, __bf16* __restrict__ dst, int nmat) {
    constexpr int NTILE = NOUT / 16;
    const int per = KD * NOUT;
    const long f = (long)blockIdx.x * blockDim.x + threadIdx.x;
    if (f >= (long)nmat * per) return;
    const int mat = (int)(f / per);
    const int r   = (int)(f - (long)mat * per);
    const int e    = r & 15;
    const int lane = (r >> 4) & 31;
    const int tc   = r >> 9;
    const int t    = tc % NTILE;
    const int c    = tc / NTILE;
    const int k = c * 32 + (lane >> 4) * 16 + e;
    const int n = t * 16 + (lane & 15);
    dst[f] = (__bf16)W[(size_t)mat * per + (size_t)k * NOUT + n];
}

// ---------------------------------------------------------------------------
// WMMA GEMM:  Out[M,NOUT] = A[M,KD] @ W[KD,NOUT] + bias
//   ACC:   Out += result            RESID: Out = Resid + result
// Block = 256 threads = 8 waves; each wave computes 16 rows x NOUT cols.
// ---------------------------------------------------------------------------
template <int KD, int NOUT, bool ACC, bool RESID>
__global__ __launch_bounds__(256)
void gemm_wmma(const float* __restrict__ A, const __bf16* __restrict__ Wsw,
               const float* __restrict__ bias, const float* __restrict__ Resid,
               float* __restrict__ Out, int M) {
    constexpr int NTILE = NOUT / 16;   // 16-col output tiles
    constexpr int KCH   = KD / 32;     // 32-deep K chunks per wmma

    __shared__ __attribute__((aligned(32))) __bf16 sW[KD * NOUT];

    // ---- stage pre-swizzled weights into LDS -------------------------------
#if USE_TDM && __has_builtin(__builtin_amdgcn_tensor_load_to_lds)
    if (threadIdx.x < 32) {                       // one wave drives the TDM
        const unsigned ndw = (unsigned)(KD * NOUT) / 2;     // dwords to move
        const unsigned ldsoff = (unsigned)(uintptr_t)(&sW[0]);  // LDS addr = flat[31:0]
        const unsigned long long ga = (unsigned long long)(uintptr_t)Wsw;
        v4u g0;
        g0[0] = 1u;                                         // count=1 (valid user D#)
        g0[1] = ldsoff;                                     // lds_addr
        g0[2] = (unsigned)(ga & 0xFFFFFFFFu);               // global_addr[31:0]
        g0[3] = (unsigned)((ga >> 32) & 0x01FFFFFFu)        // global_addr[56:32]
                | 0x80000000u;                              // type=2 ("image")
        v8i g1;
        g1[0] = (int)(2u << 16);                            // data_size=4B
        g1[1] = (int)((ndw & 0xFFFFu) << 16);               // tensor_dim0[15:0]
        g1[2] = (int)(((ndw >> 16) & 0xFFFFu) | (1u << 16)); // dim0[31:16], tensor_dim1=1
        g1[3] = (int)((ndw & 0xFFFFu) << 16);               // tile_dim0 = ndw
        g1[4] = 1;                                          // tile_dim1=1, tile_dim2=0
        g1[5] = (int)ndw;                                   // tensor_dim0_stride[31:0]
        g1[6] = 0;
        g1[7] = 0;
        const v4i gz4 = {0, 0, 0, 0};
        const v8i gz8 = {0, 0, 0, 0, 0, 0, 0, 0};
        __builtin_amdgcn_tensor_load_to_lds(g0, g1, gz4, gz4, gz8, 0);
        __builtin_amdgcn_s_wait_tensorcnt(0);
    }
    __syncthreads();
#else
    {
        const uint4* s128 = reinterpret_cast<const uint4*>(Wsw);
        uint4*       d128 = reinterpret_cast<uint4*>(sW);
        for (int i = threadIdx.x; i < (KD * NOUT) / 8; i += blockDim.x)
            d128[i] = s128[i];
        __syncthreads();
    }
#endif

    const int wave = threadIdx.x >> 5;
    const int lane = threadIdx.x & 31;
    const int row0 = blockIdx.x * 128 + wave * 16;
    if (row0 >= M) return;             // wave-uniform: EXEC stays all-1s (WMMA req)

    const int m    = lane & 15;
    const int half = lane >> 4;

    v8f acc[NTILE];
    const v8f vzero = {0.f, 0.f, 0.f, 0.f, 0.f, 0.f, 0.f, 0.f};
#pragma unroll
    for (int t = 0; t < NTILE; ++t) acc[t] = vzero;

#pragma unroll
    for (int c = 0; c < KCH; ++c) {
        // ---- A fragment: 16x32 bf16; lane holds one 16-float contiguous run (k = 16*half + e)
        const float* arow = A + (size_t)(row0 + m) * KD + c * 32 + half * 16;
        const float4 q0 = *reinterpret_cast<const float4*>(arow);
        const float4 q1 = *reinterpret_cast<const float4*>(arow + 4);
        const float4 q2 = *reinterpret_cast<const float4*>(arow + 8);
        const float4 q3 = *reinterpret_cast<const float4*>(arow + 12);
        v16bf af;
        af[0]  = (__bf16)q0.x; af[1]  = (__bf16)q0.y; af[2]  = (__bf16)q0.z; af[3]  = (__bf16)q0.w;
        af[4]  = (__bf16)q1.x; af[5]  = (__bf16)q1.y; af[6]  = (__bf16)q1.z; af[7]  = (__bf16)q1.w;
        af[8]  = (__bf16)q2.x; af[9]  = (__bf16)q2.y; af[10] = (__bf16)q2.z; af[11] = (__bf16)q2.w;
        af[12] = (__bf16)q3.x; af[13] = (__bf16)q3.y; af[14] = (__bf16)q3.z; af[15] = (__bf16)q3.w;
#pragma unroll
        for (int t = 0; t < NTILE; ++t) {
            // ---- B fragment: one aligned 32B LDS read (pre-swizzled)
            const v16bf bf = *reinterpret_cast<const v16bf*>(
                sW + ((c * NTILE + t) * 32 + lane) * 16);
            acc[t] = __builtin_amdgcn_wmma_f32_16x16x32_bf16(
                false, af, false, bf, (short)0, acc[t], false, false);
        }
    }

    // ---- D layout: lane col n = t*16 + l%16; VGPR j -> row (half*8 + j)
#pragma unroll
    for (int t = 0; t < NTILE; ++t) {
        const int n  = t * 16 + (lane & 15);
        const float bv = bias[n];
#pragma unroll
        for (int j = 0; j < 8; ++j) {
            const int r = row0 + half * 8 + j;
            const size_t idx = (size_t)r * NOUT + n;
            float v = acc[t][j] + bv;
            if (ACC)        Out[idx] += v;
            else if (RESID) Out[idx]  = Resid[idx] + v;
            else            Out[idx]  = v;
        }
    }
}

// ---------------------------------------------------------------------------
// BatchNorm stage 1: per-channel sum / sumsq (stats[0:64]=sum, [64:128]=sumsq)
// ---------------------------------------------------------------------------
__global__ __launch_bounds__(256)
void bn_stats(const float* __restrict__ X, int N, float* __restrict__ stats) {
    __shared__ float s1[256], s2[256];
    const int c = threadIdx.x & 63;
    const int g = threadIdx.x >> 6;
    float a = 0.f, b = 0.f;
    for (int r = blockIdx.x * 4 + g; r < N; r += gridDim.x * 4) {
        const float v = X[(size_t)r * 64 + c];
        a += v;
        b += v * v;
    }
    s1[threadIdx.x] = a;
    s2[threadIdx.x] = b;
    __syncthreads();
    if (g == 0) {
#pragma unroll
        for (int q = 1; q < 4; ++q) {
            a += s1[q * 64 + c];
            b += s2[q * 64 + c];
        }
        atomicAdd(&stats[c], a);
        atomicAdd(&stats[64 + c], b);
    }
}

// ---------------------------------------------------------------------------
// BatchNorm stage 2 + LeakyReLU(0.01):  Y = lrelu((X-m)*rsqrt(v+eps)*g + b)
// ---------------------------------------------------------------------------
__global__ __launch_bounds__(256)
void bn_act(const float* __restrict__ X, const float* __restrict__ stats,
            const float* __restrict__ gamma, const float* __restrict__ beta,
            float* __restrict__ Y, int N) {
    const size_t i = (size_t)blockIdx.x * blockDim.x + threadIdx.x;
    if (i >= (size_t)N * 64) return;
    const int c = (int)(i & 63);
    const float invN = 1.f / (float)N;
    const float mean = stats[c] * invN;
    const float var  = stats[64 + c] * invN - mean * mean;
    const float v = (X[i] - mean) * rsqrtf(var + 1e-5f) * gamma[c] + beta[c];
    Y[i] = (v >= 0.f) ? v : 0.01f * v;
}

// ---------------------------------------------------------------------------
// Scatter-add over edges: dst[edst[e]] += src[esrc[e]]  (C=64 channels)
// One thread per (edge, 4-channel group): float4 gather + 4 atomic adds.
// ---------------------------------------------------------------------------
__global__ __launch_bounds__(256)
void scatter_add(const float* __restrict__ src, const int* __restrict__ es,
                 const int* __restrict__ ed, float* __restrict__ dst, int E) {
    const long idx = (long)blockIdx.x * blockDim.x + threadIdx.x;
    if (idx >= (long)E * 16) return;
    const int e  = (int)(idx >> 4);
    const int c4 = (int)(idx & 15) << 2;
    const float4 v = *reinterpret_cast<const float4*>(src + (size_t)es[e] * 64 + c4);
    float* d = dst + (size_t)ed[e] * 64 + c4;
    atomicAdd(d + 0, v.x);
    atomicAdd(d + 1, v.y);
    atomicAdd(d + 2, v.z);
    atomicAdd(d + 3, v.w);
}

// ---------------------------------------------------------------------------
// Host orchestration
// ---------------------------------------------------------------------------
extern "C" void kernel_launch(void* const* d_in, const int* in_sizes, int n_in,
                              void* d_out, int out_size, void* d_ws, size_t ws_size,
                              hipStream_t stream) {
    const float* x_bus      = (const float*)d_in[0];
    const float* x_gen      = (const float*)d_in[1];
    const float* ri_W_bus   = (const float*)d_in[2];
    const float* ri_b_bus   = (const float*)d_in[3];
    const float* ri_W_gen   = (const float*)d_in[4];
    const float* ri_b_gen   = (const float*)d_in[5];
    const float* bn_g_bus   = (const float*)d_in[6];
    const float* bn_b_bus   = (const float*)d_in[7];
    const float* bn_g_gen   = (const float*)d_in[8];
    const float* bn_b_gen   = (const float*)d_in[9];
    const float* taps_W_bus = (const float*)d_in[10];
    const float* taps_b_bus = (const float*)d_in[11];
    const float* taps_W_gen = (const float*)d_in[12];
    const float* taps_b_gen = (const float*)d_in[13];
    const float* ro_W_bus   = (const float*)d_in[14];
    const float* ro_b_bus   = (const float*)d_in[15];
    const float* ro_W_gen   = (const float*)d_in[16];
    const float* ro_b_gen   = (const float*)d_in[17];
    const int*   e_bb       = (const int*)d_in[18];   // [2, EBB]
    const int*   e_gb_src   = (const int*)d_in[19];
    const int*   e_gb_dst   = (const int*)d_in[20];
    const int*   e_bg_src   = (const int*)d_in[21];
    const int*   e_bg_dst   = (const int*)d_in[22];
    const int* bbs = e_bb;
    const int* bbd = e_bb + EBB;

    float* out = (float*)d_out;

    // ---- workspace layout --------------------------------------------------
    // [0, 256KB): pre-swizzled bf16 weights; then f32 activation buffers.
    __bf16* wsw = (__bf16*)d_ws;
    __bf16* sw_ri_b   = wsw;                 // 2048
    __bf16* sw_ri_g   = wsw + 2048;          // 2048
    __bf16* sw_taps_b = wsw + 4096;          // 10 * 4096
    __bf16* sw_taps_g = wsw + 45056;         // 10 * 4096
    __bf16* sw_ro_b   = wsw + 86016;         // 2048
    __bf16* sw_ro_g   = wsw + 88064;         // 2048

    const size_t NBC = (size_t)NB * CC;
    const size_t NGC = (size_t)NG * CC;
    float* f = (float*)((char*)d_ws + (256 << 10));
    float* xb  = f;   f += NBC;
    float* zb  = f;   f += NBC;
    float* pb0 = f;   f += NBC;
    float* pb1 = f;   f += NBC;
    float* xg  = f;   f += NGC;
    float* zg  = f;   f += NGC;
    float* pg0 = f;   f += NGC;
    float* pg1 = f;   f += NGC;
    float* st_b = f;  f += 128;
    float* st_g = f;  f += 128;

    const dim3 blk(256);
    const dim3 gB(NB / 128), gG(NG / 128);
    const dim3 gEbb((unsigned)(((long)EBB * 16 + 255) / 256));
    const dim3 gEgb((unsigned)(((long)EGB * 16 + 255) / 256));
    const dim3 gEbg((unsigned)(((long)EBG * 16 + 255) / 256));
    const dim3 gNBel((unsigned)((NBC + 255) / 256));
    const dim3 gNGel((unsigned)((NGC + 255) / 256));

    // ---- swizzle all weights into fragment order (bf16) --------------------
    swizzle_w<CIN, CC><<<dim3(8),  blk, 0, stream>>>(ri_W_bus,   sw_ri_b,   1);
    swizzle_w<CIN, CC><<<dim3(8),  blk, 0, stream>>>(ri_W_gen,   sw_ri_g,   1);
    swizzle_w<CC,  CC><<<dim3(160), blk, 0, stream>>>(taps_W_bus, sw_taps_b, NL * NT);
    swizzle_w<CC,  CC><<<dim3(160), blk, 0, stream>>>(taps_W_gen, sw_taps_g, NL * NT);
    swizzle_w<CC, COUT><<<dim3(8), blk, 0, stream>>>(ro_W_bus,   sw_ro_b,   1);
    swizzle_w<CC, COUT><<<dim3(8), blk, 0, stream>>>(ro_W_gen,   sw_ro_g,   1);

    // ---- readin: x @ W(32x64) + b ------------------------------------------
    gemm_wmma<CIN, CC, false, false><<<gB, blk, 0, stream>>>(x_bus, sw_ri_b, ri_b_bus, nullptr, xb, NB);
    gemm_wmma<CIN, CC, false, false><<<gG, blk, 0, stream>>>(x_gen, sw_ri_g, ri_b_gen, nullptr, xg, NG);

    float* xb_p = xb; float* zb_p = zb;
    float* xg_p = xg; float* zg_p = zg;

    for (int l = 0; l < NL; ++l) {
        // ---- BatchNorm stats (st_b/st_g contiguous -> single memset)
        (void)hipMemsetAsync(st_b, 0, 256 * sizeof(float), stream);
        bn_stats<<<dim3(256), blk, 0, stream>>>(xb_p, NB, st_b);
        bn_stats<<<dim3(256), blk, 0, stream>>>(xg_p, NG, st_g);
        bn_act<<<gNBel, blk, 0, stream>>>(xb_p, st_b, bn_g_bus + l * CC, bn_b_bus + l * CC, pb0, NB);
        bn_act<<<gNGel, blk, 0, stream>>>(xg_p, st_g, bn_g_gen + l * CC, bn_b_gen + l * CC, pg0, NG);

        // ---- tap 0 fused with residual: z = x + y @ W[l,0] + b[l,0]
        gemm_wmma<CC, CC, false, true><<<gB, blk, 0, stream>>>(
            pb0, sw_taps_b + (size_t)(l * NT) * CC * CC, taps_b_bus + (size_t)(l * NT) * CC, xb_p, zb_p, NB);
        gemm_wmma<CC, CC, false, true><<<gG, blk, 0, stream>>>(
            pg0, sw_taps_g + (size_t)(l * NT) * CC * CC, taps_b_gen + (size_t)(l * NT) * CC, xg_p, zg_p, NG);

        float* cur_b = pb0; float* nxt_b = pb1;
        float* cur_g = pg0; float* nxt_g = pg1;
        for (int k = 1; k < NT; ++k) {
            // ---- graph shift (hetero scatter-add)
            (void)hipMemsetAsync(nxt_b, 0, NBC * sizeof(float), stream);
            (void)hipMemsetAsync(nxt_g, 0, NGC * sizeof(float), stream);
            scatter_add<<<gEbb, blk, 0, stream>>>(cur_b, bbs, bbd, nxt_b, EBB);
            scatter_add<<<gEgb, blk, 0, stream>>>(cur_g, e_gb_src, e_gb_dst, nxt_b, EGB);
            scatter_add<<<gEbg, blk, 0, stream>>>(cur_b, e_bg_src, e_bg_dst, nxt_g, EBG);
            // ---- z += c @ W[l,k] + b[l,k]
            gemm_wmma<CC, CC, true, false><<<gB, blk, 0, stream>>>(
                nxt_b, sw_taps_b + (size_t)(l * NT + k) * CC * CC,
                taps_b_bus + (size_t)(l * NT + k) * CC, nullptr, zb_p, NB);
            gemm_wmma<CC, CC, true, false><<<gG, blk, 0, stream>>>(
                nxt_g, sw_taps_g + (size_t)(l * NT + k) * CC * CC,
                taps_b_gen + (size_t)(l * NT + k) * CC, nullptr, zg_p, NG);
            float* t;
            t = cur_b; cur_b = nxt_b; nxt_b = t;
            t = cur_g; cur_g = nxt_g; nxt_g = t;
        }
        // ---- residual already folded into z: new x is z; recycle old x as z
        float* t;
        t = xb_p; xb_p = zb_p; zb_p = t;
        t = xg_p; xg_p = zg_p; zg_p = t;
    }

    // ---- readout: x @ W(64x32) + b -> d_out (bus then gen, flat) -----------
    gemm_wmma<CC, COUT, false, false><<<gB, blk, 0, stream>>>(xb_p, sw_ro_b, ro_b_bus, nullptr, out, NB);
    gemm_wmma<CC, COUT, false, false><<<gG, blk, 0, stream>>>(xg_p, sw_ro_g, ro_b_gen, nullptr,
                                                              out + (size_t)NB * COUT, NG);
}
